// SpatioTemporalGNN_16003048145200
// MI455X (gfx1250) — matrix operand order repaired and checked by reading, not compiled, not run on hardware
//
#include <hip/hip_runtime.h>
#include <hip/hip_bf16.h>
#include <cstdint>

typedef __attribute__((ext_vector_type(16))) _Float16 v16h;
typedef __attribute__((ext_vector_type(8)))  float    v8f;
typedef _Float16 half_t;

#define WMMA_F32_F16(a, b, c) \
  __builtin_amdgcn_wmma_f32_16x16x32_f16(false, (a), false, (b), (short)0, (c), false, false)

// ---------------- problem constants ----------------
#define BB   256
#define TT   256
#define NN   8
#define FIN  5
#define EE   24      // NE_RAND(16) + N(8) self loops
#define H1C  64      // HEADS1*GAT_H
#define C2   32      // GAT_H
#define GH   128     // GRU hidden
#define G3H  384     // 3*GRU_H
#define OD   8

__constant__ float c_node_pos[8] = {0.f, 1.f, 2.f, 2.f, 1.f, 2.f, 1.f, 2.f};

// =====================================================================
// Kernel 1: GAT front-end.  128 threads = 16 graphs x 8 nodes.
// LDS: weights + per-graph scratch.  Dense 8x8 attention matrix per head.
// =====================================================================
#define GPB 16
// per-graph LDS layout (floats)
#define PG_BUFA   0      // [8][64] layer1 xp / layer2 xp
#define PG_BUFB   512    // [8][64] h1 / h2
#define PG_ES     1024   // [8][2]
#define PG_ED     1040   // [8][2]
#define PG_P      1056   // [2][8][8] attention matrices
#define PG_SC     1184   // [8] pooling scores
#define PG_STRIDE 1192
// weight region (floats)
#define W_W1   0
#define W_B1   320
#define W_AS1  384
#define W_AD1  448
#define W_W2   512
#define W_B2   2560
#define W_AS2  2592
#define W_AD2  2624
#define W_PW   2656
#define W_NE   2688
#define W_PSW  2728
#define W_PSB  2733
#define W_PB   2738
#define W_SRC  2740   // 24 ints
#define W_DST  2764   // 24 ints
#define W_TOT  2788
#define K1_LDS_BYTES ((W_TOT + GPB * PG_STRIDE) * 4)

__global__ __launch_bounds__(128) void k1_gat_frontend(
    const float* __restrict__ x, const int* __restrict__ esrc, const int* __restrict__ edst,
    const float* __restrict__ node_emb, const float* __restrict__ pos_w, const float* __restrict__ pos_b,
    const float* __restrict__ W1, const float* __restrict__ as1, const float* __restrict__ ad1,
    const float* __restrict__ b1v,
    const float* __restrict__ W2, const float* __restrict__ as2, const float* __restrict__ ad2,
    const float* __restrict__ b2v,
    const float* __restrict__ pwv, const float* __restrict__ pbv,
    float* __restrict__ pooled)
{
  extern __shared__ float smf[];
  const int tid = threadIdx.x;

  for (int i = tid; i < 320;  i += 128) smf[W_W1 + i] = W1[i];
  for (int i = tid; i < 64;   i += 128) { smf[W_B1+i]=b1v[i]; smf[W_AS1+i]=as1[i]; smf[W_AD1+i]=ad1[i]; }
  for (int i = tid; i < 2048; i += 128) smf[W_W2 + i] = W2[i];
  if (tid < 32) { smf[W_B2+tid]=b2v[tid]; smf[W_AS2+tid]=as2[tid]; smf[W_AD2+tid]=ad2[tid]; smf[W_PW+tid]=pwv[tid]; }
  if (tid < 40) smf[W_NE+tid] = node_emb[tid];
  if (tid < 5)  { smf[W_PSW+tid]=pos_w[tid]; smf[W_PSB+tid]=pos_b[tid]; }
  if (tid == 0) smf[W_PB] = pbv[0];
  int* ssrc = (int*)(smf + W_SRC);
  int* sdst = (int*)(smf + W_DST);
  if (tid < EE) { ssrc[tid] = esrc[tid]; sdst[tid] = edst[tid]; }
  __syncthreads();

  const int gl = tid >> 3;
  const int n  = tid & 7;
  const int g  = blockIdx.x * GPB + gl;
  float* gb = smf + W_TOT + gl * PG_STRIDE;

  // ---- phase 1: embeddings + layer-1 projection + attention logits ----
  float xg[FIN];
  const float np = c_node_pos[n];
#pragma unroll
  for (int f = 0; f < FIN; ++f)
    xg[f] = x[(size_t)(g * NN + n) * FIN + f] + smf[W_NE + n*FIN + f]
            + np * smf[W_PSW + f] + smf[W_PSB + f];
  float es0 = 0.f, es1 = 0.f, ed0 = 0.f, ed1 = 0.f;
#pragma unroll
  for (int o = 0; o < H1C; ++o) {
    float v = 0.f;
#pragma unroll
    for (int f = 0; f < FIN; ++f) v += smf[W_W1 + o*FIN + f] * xg[f];
    gb[PG_BUFA + n*64 + o] = v;
    if (o < 32) { es0 += v * smf[W_AS1+o]; ed0 += v * smf[W_AD1+o]; }
    else        { es1 += v * smf[W_AS1+o]; ed1 += v * smf[W_AD1+o]; }
  }
  gb[PG_ES + n*2 + 0] = es0; gb[PG_ES + n*2 + 1] = es1;
  gb[PG_ED + n*2 + 0] = ed0; gb[PG_ED + n*2 + 1] = ed1;
  __syncthreads();

  // ---- phase 2: layer-1 segment softmax -> dense P, aggregate, ELU ----
  {
    float em0 = -1e30f, em1 = -1e30f;
    for (int e = 0; e < EE; ++e) {
      if (sdst[e] == n) {
        const int s = ssrc[e];
        float v0 = gb[PG_ES + s*2 + 0] + ed0; v0 = v0 > 0.f ? v0 : 0.2f * v0;
        float v1 = gb[PG_ES + s*2 + 1] + ed1; v1 = v1 > 0.f ? v1 : 0.2f * v1;
        em0 = fmaxf(em0, v0); em1 = fmaxf(em1, v1);
      }
    }
#pragma unroll
    for (int s = 0; s < 8; ++s) { gb[PG_P + n*8 + s] = 0.f; gb[PG_P + 64 + n*8 + s] = 0.f; }
    float dn0 = 0.f, dn1 = 0.f;
    for (int e = 0; e < EE; ++e) {
      if (sdst[e] == n) {
        const int s = ssrc[e];
        float v0 = gb[PG_ES + s*2 + 0] + ed0; v0 = v0 > 0.f ? v0 : 0.2f * v0;
        float v1 = gb[PG_ES + s*2 + 1] + ed1; v1 = v1 > 0.f ? v1 : 0.2f * v1;
        const float a0 = expf(v0 - em0), a1 = expf(v1 - em1);
        dn0 += a0; dn1 += a1;
        gb[PG_P + n*8 + s] += a0;
        gb[PG_P + 64 + n*8 + s] += a1;
      }
    }
    const float r0 = 1.f / dn0, r1 = 1.f / dn1;
#pragma unroll
    for (int s = 0; s < 8; ++s) { gb[PG_P + n*8 + s] *= r0; gb[PG_P + 64 + n*8 + s] *= r1; }
#pragma unroll
    for (int o = 0; o < H1C; ++o) {
      const int h = o >> 5;
      float acc = 0.f;
#pragma unroll
      for (int s = 0; s < 8; ++s) acc += gb[PG_P + h*64 + n*8 + s] * gb[PG_BUFA + s*64 + o];
      acc += smf[W_B1 + o];
      gb[PG_BUFB + n*64 + o] = acc > 0.f ? acc : expf(acc) - 1.f;   // ELU
    }
  }
  __syncthreads();

  // ---- phase 3: layer-2 projection + logits ----
  float es2 = 0.f, ed2 = 0.f;
#pragma unroll
  for (int c = 0; c < C2; ++c) {
    float v = 0.f;
#pragma unroll
    for (int o = 0; o < H1C; ++o) v += smf[W_W2 + c*64 + o] * gb[PG_BUFB + n*64 + o];
    gb[PG_BUFA + n*64 + c] = v;
    es2 += v * smf[W_AS2 + c];
    ed2 += v * smf[W_AD2 + c];
  }
  gb[PG_ES + n*2] = es2; gb[PG_ED + n*2] = ed2;
  __syncthreads();

  // ---- phase 4: layer-2 attention + ELU + pooling score ----
  {
    float em = -1e30f;
    for (int e = 0; e < EE; ++e) {
      if (sdst[e] == n) {
        float v = gb[PG_ES + ssrc[e]*2] + ed2; v = v > 0.f ? v : 0.2f * v;
        em = fmaxf(em, v);
      }
    }
#pragma unroll
    for (int s = 0; s < 8; ++s) gb[PG_P + n*8 + s] = 0.f;
    float dn = 0.f;
    for (int e = 0; e < EE; ++e) {
      if (sdst[e] == n) {
        const int s = ssrc[e];
        float v = gb[PG_ES + s*2] + ed2; v = v > 0.f ? v : 0.2f * v;
        const float a = expf(v - em);
        dn += a; gb[PG_P + n*8 + s] += a;
      }
    }
    const float rr = 1.f / dn;
#pragma unroll
    for (int s = 0; s < 8; ++s) gb[PG_P + n*8 + s] *= rr;
    float sc = smf[W_PB];
#pragma unroll
    for (int c = 0; c < C2; ++c) {
      float acc = 0.f;
#pragma unroll
      for (int s = 0; s < 8; ++s) acc += gb[PG_P + n*8 + s] * gb[PG_BUFA + s*64 + c];
      acc += smf[W_B2 + c];
      acc = acc > 0.f ? acc : expf(acc) - 1.f;
      gb[PG_BUFB + n*64 + c] = acc;
      sc += acc * smf[W_PW + c];
    }
    gb[PG_SC + n] = sc;
  }
  __syncthreads();

  // ---- phase 5: softmax pooling over the 8 nodes ----
  float m = gb[PG_SC + 0];
#pragma unroll
  for (int s = 1; s < 8; ++s) m = fmaxf(m, gb[PG_SC + s]);
  float wn[8], wsum = 0.f;
#pragma unroll
  for (int s = 0; s < 8; ++s) { wn[s] = expf(gb[PG_SC + s] - m); wsum += wn[s]; }
  const float rw = 1.f / wsum;
#pragma unroll
  for (int k = 0; k < 4; ++k) {
    const int c = n + 8 * k;
    float acc = 0.f;
#pragma unroll
    for (int s = 0; s < 8; ++s) acc += wn[s] * gb[PG_BUFB + s*64 + c];
    pooled[(size_t)g * C2 + c] = acc * rw;
  }
}

// =====================================================================
// Kernel 2: GI = pooled @ wih.T   [65536,32]x[32,384] via WMMA f16-split
// grid (4096, 3), block 256 (8 waves); wave w -> N-tile blockIdx.y*8+w
// =====================================================================
__global__ __launch_bounds__(256) void k2_gi_gemm(
    const float* __restrict__ pooled, const float* __restrict__ wih, float* __restrict__ GI)
{
  const int tid = threadIdx.x;
  const int w = tid >> 5, L = tid & 31;
  const int nt = blockIdx.y * 8 + w;
  const int mt = blockIdx.x;
  const int kb8 = (L >> 4) << 3;

  // A fragment: pooled rows, K = 32 channels (hi/lo f16 split)
  v16h Ah{}, Al{};
  {
    const float* pr = pooled + (size_t)(mt * 16 + (L & 15)) * C2;
#pragma unroll
    for (int e = 0; e < 16; ++e) {
      const int K = 2 * kb8 + e;
      const float v = pr[K];
      const half_t hi = (half_t)v;
      Ah[e] = hi; Al[e] = (half_t)(v - (float)hi);
    }
  }
  // B fragment: wih.T  -> B[k][n] = wih[n*32+k]; per lane 16 contiguous k
  v16h Bh{}, Bl{};
  {
    const int N = nt * 16 + (L & 15);
    const float* wr = wih + N * C2 + ((L >> 4) << 4);
#pragma unroll
    for (int e = 0; e < 16; ++e) {
      const float v = wr[e];
      const half_t hi = (half_t)v;
      Bh[e] = hi; Bl[e] = (half_t)(v - (float)hi);
    }
  }
  v8f acc = {0.f,0.f,0.f,0.f,0.f,0.f,0.f,0.f};
  acc = WMMA_F32_F16(Al, Bh, acc);
  acc = WMMA_F32_F16(Ah, Bl, acc);
  acc = WMMA_F32_F16(Ah, Bh, acc);
  const int N = nt * 16 + (L & 15);
#pragma unroll
  for (int jj = 0; jj < 8; ++jj)
    GI[(size_t)(mt * 16 + jj + kb8) * G3H + N] = acc[jj];
}

// =====================================================================
// Kernel 3: GRU recurrence (+ final FC).  16 blocks x 256 threads,
// 16 batch rows per block.  whh/wih pre-split hi/lo f16, packed in LDS
// in exact B-fragment order (conflict-free ds_load_b128).
// =====================================================================
#define HSTR  133   // h-state LDS stride (dwords): 5*M+K conflict-free
#define GHSTR 388   // gh/gi LDS stride (dwords): disjoint bank halves
#define L_WHH_HI 0u
#define L_WHH_LO 98304u
#define L_WIH_HI 196608u
#define L_WIH_LO 221184u
#define L_H      245760u
#define L_GH     254272u
#define L_GI     279104u
#define K3_LDS_BYTES 303936u

__global__ __launch_bounds__(256) void k3_gru(
    const float* __restrict__ pooled, const float* __restrict__ GI,
    const float* __restrict__ wih, const float* __restrict__ whh,
    const float* __restrict__ bih, const float* __restrict__ bhh,
    const float* __restrict__ fcw, const float* __restrict__ fcb,
    float* __restrict__ out, const int use_gi)
{
  extern __shared__ char k3s[];
  half_t* whh_hi = (half_t*)(k3s + L_WHH_HI);
  half_t* whh_lo = (half_t*)(k3s + L_WHH_LO);
  half_t* wih_hi = (half_t*)(k3s + L_WIH_HI);
  half_t* wih_lo = (half_t*)(k3s + L_WIH_LO);
  float*  sh  = (float*)(k3s + L_H);
  float*  sgh = (float*)(k3s + L_GH);
  float*  sgi = (float*)(k3s + L_GI);
  const int tid = threadIdx.x;

  // pack whh into B-fragment order, hi/lo f16 split
  for (int idx = tid; idx < 24*4*32*16; idx += 256) {
    const int e = idx & 15, lane = (idx >> 4) & 31, kc = (idx >> 9) & 3, nt = idx >> 11;
    const int nrow = nt * 16 + (lane & 15);
    const int k = kc * 32 + ((lane >> 4) << 4) + e;
    const float v = whh[nrow * GH + k];
    const half_t hi = (half_t)v;
    whh_hi[idx] = hi; whh_lo[idx] = (half_t)(v - (float)hi);
  }
  if (!use_gi) {
    for (int idx = tid; idx < 24*32*16; idx += 256) {
      const int e = idx & 15, lane = (idx >> 4) & 31, nt = idx >> 9;
      const int nrow = nt * 16 + (lane & 15);
      const int k = ((lane >> 4) << 4) + e;
      const float v = wih[nrow * C2 + k];
      const half_t hi = (half_t)v;
      wih_hi[idx] = hi; wih_lo[idx] = (half_t)(v - (float)hi);
    }
  }
  for (int i = tid; i < 16 * HSTR; i += 256) sh[i] = 0.f;   // h0 = 0
  __syncthreads();

  const int b0 = blockIdx.x * 16;
  const int w = tid >> 5, L = tid & 31;
  const int Mrow = L & 15;
  const int kb8 = (L >> 4) << 3;
  const int j = tid & 127;
  const int rhalf = tid >> 7;
  const float bihr = bih[j], bihz = bih[j+128], bihn = bih[j+256];
  const float bhhr = bhh[j], bhhz = bhh[j+128], bhhn = bhh[j+256];

  for (int t = 0; t < TT; ++t) {
    // GI loads issued early -> overlap with the WMMA chain
    float gir[8], giz[8], gin[8];
    if (use_gi) {
#pragma unroll
      for (int i = 0; i < 8; ++i) {
        const int b = b0 + rhalf * 8 + i;
        const float* gp = GI + ((size_t)b * TT + t) * G3H;
        gir[i] = gp[j]; giz[i] = gp[j+128]; gin[i] = gp[j+256];
      }
      if (t + 1 < TT)
        __builtin_prefetch(GI + ((size_t)(b0 + rhalf*8) * TT + t + 1) * G3H + j, 0, 1);
    }
    // A2 fragment (pooled) for on-the-fly gi path
    v16h A2h{}, A2l{};
    if (!use_gi) {
      const float* pp = pooled + ((size_t)(b0 + Mrow) * TT + t) * C2;
#pragma unroll
      for (int e = 0; e < 16; ++e) {
        const int K = 2 * kb8 + e;
        const float v = pp[K];
        const half_t hi = (half_t)v;
        A2h[e] = hi; A2l[e] = (half_t)(v - (float)hi);
      }
    }
    // A fragments from the hidden state (4 K-chunks), hi/lo split
    v16h Ah[4], Al[4];
#pragma unroll
    for (int kc = 0; kc < 4; ++kc) {
#pragma unroll
      for (int e = 0; e < 16; ++e) {
        const int K = kc * 32 + 2 * kb8 + e;
        const float v = sh[Mrow * HSTR + K];
        const half_t hi = (half_t)v;
        Ah[kc][e] = hi; Al[kc][e] = (half_t)(v - (float)hi);
      }
    }
    // gh = h @ whh.T (+ gi on the fly if needed): wave handles 3 N-tiles
#pragma unroll
    for (int q = 0; q < 3; ++q) {
      const int nt = w * 3 + q;
      v8f acc = {0.f,0.f,0.f,0.f,0.f,0.f,0.f,0.f};
#pragma unroll
      for (int kc = 0; kc < 4; ++kc) {
        const v16h Bh = *(const v16h*)(whh_hi + ((size_t)(nt*4 + kc)*32 + L) * 16);
        const v16h Bl = *(const v16h*)(whh_lo + ((size_t)(nt*4 + kc)*32 + L) * 16);
        acc = WMMA_F32_F16(Al[kc], Bh, acc);
        acc = WMMA_F32_F16(Ah[kc], Bl, acc);
        acc = WMMA_F32_F16(Ah[kc], Bh, acc);
      }
#pragma unroll
      for (int jj = 0; jj < 8; ++jj)
        sgh[(jj + kb8) * GHSTR + nt * 16 + Mrow] = acc[jj];
      if (!use_gi) {
        v8f acc2 = {0.f,0.f,0.f,0.f,0.f,0.f,0.f,0.f};
        const v16h Bh2 = *(const v16h*)(wih_hi + ((size_t)nt*32 + L) * 16);
        const v16h Bl2 = *(const v16h*)(wih_lo + ((size_t)nt*32 + L) * 16);
        acc2 = WMMA_F32_F16(A2l, Bh2, acc2);
        acc2 = WMMA_F32_F16(A2h, Bl2, acc2);
        acc2 = WMMA_F32_F16(A2h, Bh2, acc2);
#pragma unroll
        for (int jj = 0; jj < 8; ++jj)
          sgi[(jj + kb8) * GHSTR + nt * 16 + Mrow] = acc2[jj];
      }
    }
    __syncthreads();   // gh/gi ready; all h reads done

    // gate math: thread owns (j, rows rhalf*8 .. rhalf*8+7)
#pragma unroll
    for (int i = 0; i < 8; ++i) {
      const int rb = rhalf * 8 + i;
      float gr, gz, gn2;
      if (use_gi) { gr = gir[i]; gz = giz[i]; gn2 = gin[i]; }
      else {
        gr  = sgi[rb*GHSTR + j];
        gz  = sgi[rb*GHSTR + j + 128];
        gn2 = sgi[rb*GHSTR + j + 256];
      }
      const float hr = sgh[rb*GHSTR + j]       + bhhr;
      const float hz = sgh[rb*GHSTR + j + 128] + bhhz;
      const float hn = sgh[rb*GHSTR + j + 256] + bhhn;
      const float r  = 1.f / (1.f + expf(-(gr + bihr + hr)));
      const float z  = 1.f / (1.f + expf(-(gz + bihz + hz)));
      const float nn = tanhf(gn2 + bihn + r * hn);
      const float hp = sh[rb*HSTR + j];
      sh[rb*HSTR + j] = (1.f - z) * nn + z * hp;
    }
    __syncthreads();   // h ready for next step
  }

  // fused FC: out = h_T @ fc_w.T + fc_b
  if (tid < 128) {
    const int rb = tid >> 3, od = tid & 7;
    float acc = fcb[od];
    for (int k = 0; k < GH; ++k) acc += sh[rb*HSTR + k] * fcw[od*GH + k];
    out[(b0 + rb) * OD + od] = acc;
  }
}

// =====================================================================
extern "C" void kernel_launch(void* const* d_in, const int* in_sizes, int n_in,
                              void* d_out, int out_size, void* d_ws, size_t ws_size,
                              hipStream_t stream) {
  (void)in_sizes; (void)n_in; (void)out_size;
  const float* x        = (const float*)d_in[0];
  const int*   esrc     = (const int*)  d_in[1];
  const int*   edst     = (const int*)  d_in[2];
  const float* node_emb = (const float*)d_in[3];
  const float* pos_w    = (const float*)d_in[4];
  const float* pos_b    = (const float*)d_in[5];
  const float* W1       = (const float*)d_in[6];
  const float* as1      = (const float*)d_in[7];
  const float* ad1      = (const float*)d_in[8];
  const float* b1v      = (const float*)d_in[9];
  const float* W2       = (const float*)d_in[10];
  const float* as2      = (const float*)d_in[11];
  const float* ad2      = (const float*)d_in[12];
  const float* b2v      = (const float*)d_in[13];
  const float* pool_w   = (const float*)d_in[14];
  const float* pool_b   = (const float*)d_in[15];
  const float* gru_wih  = (const float*)d_in[16];
  const float* gru_whh  = (const float*)d_in[17];
  const float* gru_bih  = (const float*)d_in[18];
  const float* gru_bhh  = (const float*)d_in[19];
  const float* fc_w     = (const float*)d_in[20];
  const float* fc_b     = (const float*)d_in[21];
  float* out = (float*)d_out;

  const size_t G = (size_t)BB * TT;                  // 65536 graphs
  float* pooled = (float*)d_ws;                      // [G,32]   8 MB
  const size_t pooled_bytes = G * C2 * sizeof(float);
  const size_t gi_bytes     = G * G3H * sizeof(float);  // 96 MB
  float* GI = (float*)((char*)d_ws + pooled_bytes);
  const int use_gi = (ws_size >= pooled_bytes + gi_bytes) ? 1 : 0;

  // K1: GAT front-end -> pooled
  k1_gat_frontend<<<dim3((unsigned)(G / GPB)), dim3(128), K1_LDS_BYTES, stream>>>(
      x, esrc, edst, node_emb, pos_w, pos_b,
      W1, as1, ad1, b1v, W2, as2, ad2, b2v, pool_w, pool_b, pooled);

  // K2: hoisted gi GEMM (skipped if workspace too small; K3 then computes gi)
  if (use_gi) {
    k2_gi_gemm<<<dim3((unsigned)(G / 16), 3), dim3(256), 0, stream>>>(pooled, gru_wih, GI);
  }

  // K3: GRU recurrence + FC -> out
  k3_gru<<<dim3(BB / 16), dim3(256), K3_LDS_BYTES, stream>>>(
      pooled, GI, gru_wih, gru_whh, gru_bih, gru_bhh, fc_w, fc_b, out, use_gi);
}
